// SS2D_73538430042683
// MI455X (gfx1250) — matrix operand order repaired and checked by reading, not compiled, not run on hardware
//
#include <hip/hip_runtime.h>

typedef __attribute__((ext_vector_type(16))) __bf16 v16bf;
typedef __attribute__((ext_vector_type(8)))  __bf16 v8bf;
typedef __attribute__((ext_vector_type(8)))  float  v8f;

#define B_SZ   8
#define DM     192
#define DIN    384
#define HH     64
#define WW     64
#define LL     4096
#define KDIR   4
#define RR     12
#define NN     16
#define CPROJ  44
#define CPAD   48
#define NCHUNK 32
#define CHUNK  128
#define TOTELT ((size_t)B_SZ * DIN * LL)   // 12,582,912

// ---------------- workspace layout (bytes) ----------------
#define OFF_H0   ((size_t)0)           // h0 f32 [b][l][d]; reused later as y_comb
#define OFF_HT   ((size_t)50331648)    // hT f32 [b][l][d] conv output (scan u reads)
#define OFF_SBF  ((size_t)100663296)   // bf16: hr[b][l][d] then ht[b][lt][d]; front reused as yn
#define OFF_G    ((size_t)150994944)   // G f32 [b*4+k][48][L]
#define OFF_P    ((size_t)176160768)   // P f32 chunk products (also xlb bf16 early)
#define OFF_Q    ((size_t)201326592)   // Q f32 chunk zero-state results
#define OFF_WBF  ((size_t)226492416)   // bf16 weights: in(384x192) xp(4x48x384) out(192x384)
// total: 226,934,784 bytes

#define WBF_IN_ELT  (DIN * DM)          // 73728
#define WBF_XP_ELT  (KDIR * CPAD * DIN) // 73728
#define WBF_OUT_ELT (DM * DIN)          // 73728

// ---------------- WMMA: one A 16-row tile x four 16-col tiles ----------------
// A: bf16 row-major [M][Kc] (lda = Kc), fully padded, unconditional loads.
// B: bf16 l-major: element(k,n) at n*ldb + k  -> 16 contiguous bf16 per lane.
// Software-pipelined: iteration i+1's fragment loads are issued before
// iteration i's WMMAs consume their fragments, so the s_wait_loadcnt before
// each WMMA group has a full iteration of slack instead of waiting to zero.
__device__ inline void wmma_quad(const __bf16* __restrict__ A, int lda,
                                 const __bf16* __restrict__ Bm, int ldb,
                                 int m0, int n0, int Kc, v8f acc[4]) {
  const int lane = threadIdx.x & 31;
  const int half = lane >> 4;
  const int lm   = lane & 15;
  const __bf16* arow = A + (size_t)(m0 + lm) * lda + half * 16;
  const __bf16* bp0 = Bm + (size_t)(n0 +  0 + lm) * ldb + half * 16;
  const __bf16* bp1 = Bm + (size_t)(n0 + 16 + lm) * ldb + half * 16;
  const __bf16* bp2 = Bm + (size_t)(n0 + 32 + lm) * ldb + half * 16;
  const __bf16* bp3 = Bm + (size_t)(n0 + 48 + lm) * ldb + half * 16;
  // preload k0 = 0 fragments
  v8bf a0 = *(const v8bf*)(arow);
  v8bf a1 = *(const v8bf*)(arow + 8);
  v16bf af = __builtin_shufflevector(a0, a1, 0, 1, 2, 3, 4, 5, 6, 7,
                                     8, 9, 10, 11, 12, 13, 14, 15);
  v16bf b0 = *(const v16bf*)(bp0);
  v16bf b1 = *(const v16bf*)(bp1);
  v16bf b2 = *(const v16bf*)(bp2);
  v16bf b3 = *(const v16bf*)(bp3);
  for (int k0 = 0; k0 < Kc; k0 += 32) {
    v16bf afn = {}, b0n = {}, b1n = {}, b2n = {}, b3n = {};
    const int kn = k0 + 32;
    if (kn < Kc) {           // statically resolved after inlining + full unroll
      v8bf a0n = *(const v8bf*)(arow + kn);
      v8bf a1n = *(const v8bf*)(arow + kn + 8);
      afn = __builtin_shufflevector(a0n, a1n, 0, 1, 2, 3, 4, 5, 6, 7,
                                    8, 9, 10, 11, 12, 13, 14, 15);
      b0n = *(const v16bf*)(bp0 + kn);
      b1n = *(const v16bf*)(bp1 + kn);
      b2n = *(const v16bf*)(bp2 + kn);
      b3n = *(const v16bf*)(bp3 + kn);
    }
    acc[0] = __builtin_amdgcn_wmma_f32_16x16x32_bf16(false, af, false, b0,
                                                     (short)0, acc[0], false, false);
    acc[1] = __builtin_amdgcn_wmma_f32_16x16x32_bf16(false, af, false, b1,
                                                     (short)0, acc[1], false, false);
    acc[2] = __builtin_amdgcn_wmma_f32_16x16x32_bf16(false, af, false, b2,
                                                     (short)0, acc[2], false, false);
    acc[3] = __builtin_amdgcn_wmma_f32_16x16x32_bf16(false, af, false, b3,
                                                     (short)0, acc[3], false, false);
    af = afn; b0 = b0n; b1 = b1n; b2 = b2n; b3 = b3n;
  }
}

// C row-major [m][n] (ldc = row stride)
__device__ inline void store_quad_rm(float* __restrict__ C, int ldc, int m0, int n0,
                                     const v8f acc[4]) {
  const int lane = threadIdx.x & 31;
  const int half = lane >> 4;
  const int lm   = lane & 15;
  const int crow = m0 + half * 8;
#pragma unroll
  for (int nb = 0; nb < 4; ++nb)
#pragma unroll
    for (int r = 0; r < 8; ++r)
      C[(size_t)(crow + r) * ldc + n0 + nb * 16 + lm] = acc[nb][r];
}

// C transposed [n][m] (ldc = DIN): 8 contiguous floats per lane per tile
__device__ inline void store_quad_tr(float* __restrict__ C, int ldc, int m0, int n0,
                                     const v8f acc[4]) {
  const int lane = threadIdx.x & 31;
  const int half = lane >> 4;
  const int lm   = lane & 15;
#pragma unroll
  for (int nb = 0; nb < 4; ++nb) {
    float* p = C + (size_t)(n0 + nb * 16 + lm) * ldc + m0 + half * 8;
#pragma unroll
    for (int r = 0; r < 8; ++r) p[r] = acc[nb][r];
  }
}

// ---------------- prep: weights -> bf16 (x_proj padded 44->48) ----------------
__global__ void k_prep_w(const float* __restrict__ ipw, const float* __restrict__ xpw,
                         const float* __restrict__ opw, __bf16* __restrict__ wbf) {
  int idx = blockIdx.x * blockDim.x + threadIdx.x;
  if (idx < WBF_IN_ELT) {
    wbf[idx] = (__bf16)ipw[idx];
  } else if (idx < WBF_IN_ELT + WBF_XP_ELT) {
    int i = idx - WBF_IN_ELT;
    int k = i / (CPAD * DIN);
    int rem = i - k * (CPAD * DIN);
    int c = rem / DIN;
    int d = rem - c * DIN;
    float v = (c < CPROJ) ? xpw[((size_t)k * CPROJ + c) * DIN + d] : 0.0f;
    wbf[WBF_IN_ELT + i] = (__bf16)v;
  } else if (idx < WBF_IN_ELT + WBF_XP_ELT + WBF_OUT_ELT) {
    int i = idx - WBF_IN_ELT - WBF_XP_ELT;
    wbf[WBF_IN_ELT + WBF_XP_ELT + i] = (__bf16)opw[i];
  }
}

// ---------------- prep: x [b][c][l] f32 -> xlb [b][l][c] bf16 ----------------
__global__ void k_prep_x(const float* __restrict__ x, __bf16* __restrict__ xlb) {
  size_t idx = (size_t)blockIdx.x * blockDim.x + threadIdx.x;
  if (idx >= (size_t)B_SZ * DM * LL) return;
  int l = (int)(idx & (LL - 1));
  int c = (int)((idx >> 12) % DM);
  int b = (int)(idx / ((size_t)DM * LL));
  xlb[((size_t)b * LL + l) * DM + c] = (__bf16)x[idx];
}

// ---------------- K0: in_proj -> h0 [b][l][d] ----------------
__global__ void k_inproj(const __bf16* __restrict__ Wbf, const __bf16* __restrict__ xlb,
                         float* __restrict__ h0) {
  int mt = blockIdx.x;                                   // 0..23
  int nq = blockIdx.y * blockDim.y + threadIdx.y;        // 0..63
  int b  = blockIdx.z;
  const __bf16* Bm = xlb + (size_t)b * LL * DM;
  float* C = h0 + (size_t)b * LL * DIN;
  v8f acc[4] = {};
  wmma_quad(Wbf, DM, Bm, DM, mt * 16, nq * 64, DM, acc);
  store_quad_tr(C, DIN, mt * 16, nq * 64, acc);
}

// ---------------- K1: depthwise conv3x3 + bias + SiLU (d-fastest) -------------
__global__ void k_conv(const float* __restrict__ h0, const float* __restrict__ cw,
                       const float* __restrict__ cb, float* __restrict__ hT,
                       __bf16* __restrict__ hrbf, __bf16* __restrict__ htbf) {
  size_t idx = (size_t)blockIdx.x * blockDim.x + threadIdx.x;
  if (idx >= TOTELT) return;
  int d = (int)(idx % DIN);
  int l = (int)((idx / DIN) % LL);
  int b = (int)(idx / ((size_t)DIN * LL));
  int hh = l >> 6, w = l & 63;
  const float* in = h0 + (size_t)b * LL * DIN;
  float acc = cb[d];
#pragma unroll
  for (int ky = 0; ky < 3; ++ky) {
    int y = hh + ky - 1;
    if (y < 0 || y >= HH) continue;
#pragma unroll
    for (int kx = 0; kx < 3; ++kx) {
      int xw = w + kx - 1;
      if (xw < 0 || xw >= WW) continue;
      acc += cw[d * 9 + ky * 3 + kx] * in[(size_t)(y * WW + xw) * DIN + d];
    }
  }
  float s = acc / (1.0f + __expf(-acc));   // SiLU
  int post = w * HH + hh;                  // transposed scan order
  hT[((size_t)b * LL + l) * DIN + d]     = s;
  hrbf[((size_t)b * LL + l) * DIN + d]   = (__bf16)s;
  htbf[((size_t)b * LL + post) * DIN + d] = (__bf16)s;
}

__global__ void k_zero(float* __restrict__ p) {
  size_t idx = (size_t)blockIdx.x * blockDim.x + threadIdx.x;
  if (idx < TOTELT) p[idx] = 0.0f;
}

// ---------------- K2: x_proj  G[b,k,c,j] = sum_d Wk[c,d] s[b,p,j,d] ----------
__global__ void k_xproj(const __bf16* __restrict__ wxp, const __bf16* __restrict__ sbf,
                        float* __restrict__ G) {
  int mt = blockIdx.x;                                   // 0..2 (rows 0..47, pre-padded)
  int nq = blockIdx.y * blockDim.y + threadIdx.y;        // 0..63
  int z  = blockIdx.z;                                   // b*4+k
  int b = z >> 2, k = z & 3, p = k & 1;
  const __bf16* A  = wxp + (size_t)k * CPAD * DIN;
  const __bf16* Bm = sbf + (size_t)(p * B_SZ + b) * LL * DIN;
  float* C = G + (size_t)z * CPAD * LL;
  v8f acc[4] = {};
  wmma_quad(A, DIN, Bm, DIN, mt * 16, nq * 64, DIN, acc);
  store_quad_rm(C, LL, mt * 16, nq * 64, acc);
}

// ---------------- scan helpers ----------------
__device__ inline float softplusf(float x) {
  return (x > 20.0f) ? x : log1pf(__expf(x));
}

// ---------------- S1: per-chunk (prod a, state-from-zero) --------------------
__global__ void k_scan1(const float* __restrict__ G, const float* __restrict__ hT,
                        const float* __restrict__ A_logs, const float* __restrict__ dtw,
                        const float* __restrict__ dtb, float* __restrict__ P,
                        float* __restrict__ Q) {
  __shared__ float sG[CPROJ * CHUNK];
  int chunk = blockIdx.x;
  int bk = blockIdx.y;
  int b = bk >> 2, k = bk & 3;
  int d = threadIdx.x;                                   // 0..383
  const float* Gb = G + (size_t)bk * CPAD * LL;
  int t0 = chunk * CHUNK;
  for (int i = d; i < CPROJ * CHUNK; i += DIN) {
    int c = i / CHUNK, s = i - c * CHUNK;
    int t = t0 + s;
    int j = (k >= 2) ? (LL - 1 - t) : t;
    sG[c * CHUNK + s] = Gb[(size_t)c * LL + j];
  }
  __syncthreads();
  float Av[NN];
#pragma unroll
  for (int n = 0; n < NN; ++n) Av[n] = -__expf(A_logs[((size_t)k * DIN + d) * NN + n]);
  float wr[RR];
#pragma unroll
  for (int r = 0; r < RR; ++r) wr[r] = dtw[((size_t)k * DIN + d) * RR + r];
  float bias = dtb[k * DIN + d];
  float hs[NN], Pp[NN];
#pragma unroll
  for (int n = 0; n < NN; ++n) { hs[n] = 0.0f; Pp[n] = 1.0f; }
  const float* hTb = hT + (size_t)b * LL * DIN;
  for (int s = 0; s < CHUNK; ++s) {
    int t = t0 + s;
    int j = (k >= 2) ? (LL - 1 - t) : t;
    int pos = (k & 1) ? (((j & 63) << 6) | (j >> 6)) : j;
    float dacc = bias;
#pragma unroll
    for (int r = 0; r < RR; ++r) dacc = fmaf(wr[r], sG[r * CHUNK + s], dacc);
    float delta = softplusf(dacc);
    float u  = hTb[(size_t)pos * DIN + d];
    float du = delta * u;
#pragma unroll
    for (int n = 0; n < NN; ++n) {
      float a = __expf(delta * Av[n]);
      Pp[n] *= a;
      hs[n] = fmaf(a, hs[n], du * sG[(RR + n) * CHUNK + s]);
    }
  }
  size_t base = ((size_t)chunk * (B_SZ * KDIR * DIN) + (size_t)bk * DIN + d) * NN;
#pragma unroll
  for (int n = 0; n < NN; ++n) { P[base + n] = Pp[n]; Q[base + n] = hs[n]; }
}

// ---------------- S2: sequential chunk combine; P <- carry-in state ----------
__global__ void k_scan2(float* __restrict__ P, const float* __restrict__ Q) {
  int idx = blockIdx.x * blockDim.x + threadIdx.x;       // (b,k,d) flat
  if (idx >= B_SZ * KDIR * DIN) return;
  float Hc[NN];
#pragma unroll
  for (int n = 0; n < NN; ++n) Hc[n] = 0.0f;
  for (int c = 0; c < NCHUNK; ++c) {
    size_t base = ((size_t)c * (B_SZ * KDIR * DIN) + idx) * NN;
#pragma unroll
    for (int n = 0; n < NN; ++n) {
      float p = P[base + n], q = Q[base + n];
      P[base + n] = Hc[n];                               // carry-in for chunk c
      Hc[n] = fmaf(p, Hc[n], q);
    }
  }
}

// ---------------- S3: re-scan with carry-in, emit y (+ D*u) ------------------
__global__ void k_scan3(const float* __restrict__ G, const float* __restrict__ hT,
                        const float* __restrict__ A_logs, const float* __restrict__ dtw,
                        const float* __restrict__ dtb, const float* __restrict__ Ds,
                        const float* __restrict__ Hin, float* __restrict__ ycomb) {
  __shared__ float sG[CPROJ * CHUNK];
  int chunk = blockIdx.x;
  int bk = blockIdx.y;
  int b = bk >> 2, k = bk & 3;
  int d = threadIdx.x;
  const float* Gb = G + (size_t)bk * CPAD * LL;
  int t0 = chunk * CHUNK;
  for (int i = d; i < CPROJ * CHUNK; i += DIN) {
    int c = i / CHUNK, s = i - c * CHUNK;
    int t = t0 + s;
    int j = (k >= 2) ? (LL - 1 - t) : t;
    sG[c * CHUNK + s] = Gb[(size_t)c * LL + j];
  }
  __syncthreads();
  float Av[NN];
#pragma unroll
  for (int n = 0; n < NN; ++n) Av[n] = -__expf(A_logs[((size_t)k * DIN + d) * NN + n]);
  float wr[RR];
#pragma unroll
  for (int r = 0; r < RR; ++r) wr[r] = dtw[((size_t)k * DIN + d) * RR + r];
  float bias = dtb[k * DIN + d];
  float Dsv  = Ds[k * DIN + d];
  float hs[NN];
  size_t base = ((size_t)chunk * (B_SZ * KDIR * DIN) + (size_t)bk * DIN + d) * NN;
#pragma unroll
  for (int n = 0; n < NN; ++n) hs[n] = Hin[base + n];
  const float* hTb = hT + (size_t)b * LL * DIN;
  float* yb = ycomb + (size_t)b * LL * DIN;
  for (int s = 0; s < CHUNK; ++s) {
    int t = t0 + s;
    int j = (k >= 2) ? (LL - 1 - t) : t;
    int pos = (k & 1) ? (((j & 63) << 6) | (j >> 6)) : j;
    if (s + 16 < CHUNK) {                                 // prefetch u 16 steps ahead
      int t2 = t + 16;
      int j2 = (k >= 2) ? (LL - 1 - t2) : t2;
      int p2 = (k & 1) ? (((j2 & 63) << 6) | (j2 >> 6)) : j2;
      __builtin_prefetch(hTb + (size_t)p2 * DIN + d, 0, 1);
    }
    float dacc = bias;
#pragma unroll
    for (int r = 0; r < RR; ++r) dacc = fmaf(wr[r], sG[r * CHUNK + s], dacc);
    float delta = softplusf(dacc);
    float u  = hTb[(size_t)pos * DIN + d];
    float du = delta * u;
    float y = 0.0f;
#pragma unroll
    for (int n = 0; n < NN; ++n) {
      float a = __expf(delta * Av[n]);
      hs[n] = fmaf(a, hs[n], du * sG[(RR + n) * CHUNK + s]);
      y = fmaf(hs[n], sG[(RR + NN + n) * CHUNK + s], y);
    }
    atomicAdd(&yb[(size_t)pos * DIN + d], y + Dsv * u);   // d-contiguous atomics
  }
}

// ---------------- LN over channels; output bf16 [b][l][d] --------------------
__global__ void k_ln(const float* __restrict__ y, const float* __restrict__ lnw,
                     const float* __restrict__ lnb, __bf16* __restrict__ yn) {
  __shared__ float s1[DIN], s2[DIN];
  int bl = blockIdx.x;                                   // b*L + l
  int d  = threadIdx.x;
  float v = y[(size_t)bl * DIN + d];
  s1[d] = v; s2[d] = v * v;
  __syncthreads();
  for (int o = 192; o >= 3; o >>= 1) {
    if (d < o) { s1[d] += s1[d + o]; s2[d] += s2[d + o]; }
    __syncthreads();
  }
  float sum  = s1[0] + s1[1] + s1[2];
  float sum2 = s2[0] + s2[1] + s2[2];
  float mu  = sum * (1.0f / (float)DIN);
  float var = sum2 * (1.0f / (float)DIN) - mu * mu;
  float inv = rsqrtf(var + 1e-5f);
  yn[(size_t)bl * DIN + d] = (__bf16)((v - mu) * inv * lnw[d] + lnb[d]);
}

// ---------------- K5: out_proj  out[b,m,l] = sum_d W[m,d] yn[b,l,d] ----------
__global__ void k_outproj(const __bf16* __restrict__ Wbf, const __bf16* __restrict__ yn,
                          float* __restrict__ out) {
  int mt = blockIdx.x;                                   // 0..11
  int nq = blockIdx.y * blockDim.y + threadIdx.y;        // 0..63
  int b  = blockIdx.z;
  const __bf16* Bm = yn + (size_t)b * LL * DIN;
  float* C = out + (size_t)b * DM * LL;
  v8f acc[4] = {};
  wmma_quad(Wbf, DIN, Bm, DIN, mt * 16, nq * 64, DIN, acc);
  store_quad_rm(C, LL, mt * 16, nq * 64, acc);
}

// ---------------- launcher ----------------
extern "C" void kernel_launch(void* const* d_in, const int* in_sizes, int n_in,
                              void* d_out, int out_size, void* d_ws, size_t ws_size,
                              hipStream_t stream) {
  const float* x      = (const float*)d_in[0];
  const float* ipw    = (const float*)d_in[1];
  const float* cw     = (const float*)d_in[2];
  const float* cb     = (const float*)d_in[3];
  const float* xpw    = (const float*)d_in[4];
  const float* dtw    = (const float*)d_in[5];
  const float* dtb    = (const float*)d_in[6];
  const float* A_logs = (const float*)d_in[7];
  const float* Ds     = (const float*)d_in[8];
  const float* lnw    = (const float*)d_in[9];
  const float* lnb    = (const float*)d_in[10];
  const float* opw    = (const float*)d_in[11];

  char* ws = (char*)d_ws;
  float*  h0     = (float*)(ws + OFF_H0);
  float*  hT     = (float*)(ws + OFF_HT);
  __bf16* sbf    = (__bf16*)(ws + OFF_SBF);
  __bf16* hrbf   = sbf;
  __bf16* htbf   = sbf + (size_t)B_SZ * LL * DIN;
  float*  G      = (float*)(ws + OFF_G);
  float*  P      = (float*)(ws + OFF_P);
  float*  Q      = (float*)(ws + OFF_Q);
  __bf16* xlb    = (__bf16*)(ws + OFF_P);     // xlb dead before P is written
  __bf16* wbf    = (__bf16*)(ws + OFF_WBF);
  __bf16* wbf_in = wbf;
  __bf16* wbf_xp = wbf + WBF_IN_ELT;
  __bf16* wbf_op = wbf + WBF_IN_ELT + WBF_XP_ELT;
  float*  ycomb  = h0;                        // reuse h0 region
  __bf16* yn     = sbf;                       // reuse hr region
  float*  out    = (float*)d_out;

  const dim3 gblk(32, 4, 1);
  const int elblk = 256;
  const int elgrd = (int)((TOTELT + elblk - 1) / elblk);
  const int xgrd  = (int)(((size_t)B_SZ * DM * LL + elblk - 1) / elblk);
  const int wgrd  = (WBF_IN_ELT + WBF_XP_ELT + WBF_OUT_ELT + elblk - 1) / elblk;

  k_prep_w <<<wgrd, elblk, 0, stream>>>(ipw, xpw, opw, wbf);
  k_prep_x <<<xgrd, elblk, 0, stream>>>(x, xlb);
  k_inproj <<<dim3(DIN / 16, 16, B_SZ), gblk, 0, stream>>>(wbf_in, xlb, h0);
  k_conv   <<<elgrd, elblk, 0, stream>>>(h0, cw, cb, hT, hrbf, htbf);
  k_zero   <<<elgrd, elblk, 0, stream>>>(ycomb);
  k_xproj  <<<dim3(CPAD / 16, 16, B_SZ * KDIR), gblk, 0, stream>>>(wbf_xp, sbf, G);
  k_scan1  <<<dim3(NCHUNK, B_SZ * KDIR), DIN, 0, stream>>>(G, hT, A_logs, dtw, dtb, P, Q);
  k_scan2  <<<(B_SZ * KDIR * DIN + 255) / 256, 256, 0, stream>>>(P, Q);
  k_scan3  <<<dim3(NCHUNK, B_SZ * KDIR), DIN, 0, stream>>>(G, hT, A_logs, dtw, dtb, Ds, P, ycomb);
  k_ln     <<<B_SZ * LL, DIN, 0, stream>>>(ycomb, lnw, lnb, yn);
  k_outproj<<<dim3(DM / 16, 16, B_SZ), gblk, 0, stream>>>(wbf_op, yn, out);
}
